// MeasureNet_69166153335429
// MI455X (gfx1250) — matrix operand order repaired and checked by reading, not compiled, not run on hardware
//
#include <hip/hip_runtime.h>

// ---------------------------------------------------------------------------
// trace[d,a] = sum_{b,c} o[a,b] * o[a,c] * x[d,c,b],  o = normalize(org, dim=-1)
// Strategy:  W[d][c,a] = sum_b x[d,c,b] * o[a,b]   (batched GEMM, bf16 WMMA, f32 acc)
//            trace[d,a] = sum_c o_f32[a,c] * W[d][c,a]   (fused epilogue, f32)
// x is read from HBM exactly once (512 MB); conversion f32->bf16 is fused into
// the LDS staging path (native bf16 converts). o (1 MB bf16) stays L2-resident.
// ---------------------------------------------------------------------------

#define BATCH 32
#define TIMES 256
#define DIM   2048

typedef __attribute__((ext_vector_type(16))) __bf16          v16bf;
typedef __attribute__((ext_vector_type(8)))  __bf16          v8bf;
typedef __attribute__((ext_vector_type(8)))  float           v8f;
typedef __attribute__((ext_vector_type(4)))  float           v4f;
typedef __attribute__((ext_vector_type(8)))  unsigned short  v8u;
typedef __attribute__((ext_vector_type(16))) unsigned short  v16u;

union V16 {               // build a v16bf WMMA operand out of two 16B loads
    v16u  u;
    v8u   h[2];
    v16bf bf;
};

union V8 {                // 8 bf16 <-> 8 ushort (one 16B LDS store)
    v8bf b;
    v8u  u;
};

// ---------------------------------------------------------------------------
// Kernel 0: zero the 32x256 output (harness poisons it with 0xAA).
// ---------------------------------------------------------------------------
__global__ void zero_out_kernel(float* __restrict__ out) {
    int i = blockIdx.x * blockDim.x + threadIdx.x;
    if (i < BATCH * TIMES) out[i] = 0.0f;
}

// ---------------------------------------------------------------------------
// Kernel 1: o = org / max(||row||, 1e-12); write f32 copy (epilogue weights)
// and bf16 copy (WMMA B operand) into workspace.
// ---------------------------------------------------------------------------
__global__ void normalize_kernel(const float* __restrict__ org,
                                 float* __restrict__ o_f32,
                                 unsigned short* __restrict__ o_bf16) {
    const int row = blockIdx.x;           // 0..255
    const int tid = threadIdx.x;          // 0..255
    const float* r = org + (size_t)row * DIM;

    float ss = 0.0f;
    for (int i = tid; i < DIM; i += 256) {
        float v = r[i];
        ss += v * v;
    }
    __shared__ float red[256];
    red[tid] = ss;
    __syncthreads();
    for (int s = 128; s > 0; s >>= 1) {
        if (tid < s) red[tid] += red[tid + s];
        __syncthreads();
    }
    const float inv = 1.0f / fmaxf(sqrtf(red[0]), 1e-12f);

    for (int i = tid; i < DIM; i += 256) {
        float v = r[i] * inv;
        o_f32[(size_t)row * DIM + i] = v;
        union { __bf16 b; unsigned short u; } c;
        c.b = (__bf16)v;                   // native f32->bf16 convert
        o_bf16[(size_t)row * DIM + i] = c.u;
    }
}

// ---------------------------------------------------------------------------
// Kernel 2: fused batched GEMM + weighted-trace epilogue.
// grid  = (16 c-blocks, 32 batches), block = 512 threads (16 wave32 waves).
// Wave w owns a-tile w (16 columns of a). Each workgroup owns 128 rows of c.
// LDS: double-buffered 128x32 bf16 x-tile, rows padded by 8 halves (16B) so
// the A-fragment ds_load_b128 pattern is bank-conflict-free.
// ---------------------------------------------------------------------------
#define LDS_PITCH 40   // 32 payload halves + 8 pad halves (80B row stride)

__global__ void __launch_bounds__(512, 2)
trace_gemm_kernel(const float* __restrict__ x,
                  const float* __restrict__ o_f32,
                  const unsigned short* __restrict__ o_bf16,
                  float* __restrict__ out) {
    __shared__ __align__(16) unsigned short ldsA[2][128][LDS_PITCH];

    const int cBlk  = blockIdx.x;             // 0..15  -> c base = cBlk*128
    const int d0    = blockIdx.y;             // 0..31  batch
    const int cBase = cBlk * 128;

    const int tid   = threadIdx.x;            // 0..511
    const int wave  = tid >> 5;               // 0..15 -> a-tile
    const int lane  = tid & 31;
    const int m     = lane & 15;              // column within 16-wide tile
    const int hi    = lane >> 4;              // half-wave select

    const int aBase = wave * 16;

    // staging role: row r = tid/4, 8 consecutive k per thread
    const int sRow = tid >> 2;                // 0..127
    const int sCol = (tid & 3) * 8;           // 0,8,16,24

    const float* xrow = x + ((size_t)d0 * DIM + cBase + sRow) * DIM + sCol;

    // ---- stage k-chunk into LDS buffer buf (all 512 threads) ----
    auto stage = [&](int buf, int k0) {
        const float* gp = xrow + k0;
        v4f f0 = *(const v4f*)(gp);
        v4f f1 = *(const v4f*)(gp + 4);
        V8 h;                                  // native pk converts (RNE)
        h.b[0] = (__bf16)f0[0]; h.b[1] = (__bf16)f0[1];
        h.b[2] = (__bf16)f0[2]; h.b[3] = (__bf16)f0[3];
        h.b[4] = (__bf16)f1[0]; h.b[5] = (__bf16)f1[1];
        h.b[6] = (__bf16)f1[2]; h.b[7] = (__bf16)f1[3];
        *(v8u*)&ldsA[buf][sRow][sCol] = h.u;   // ds_store_b128
    };

    // ---- accumulators: 8 c-tiles x (16x16 f32) = 64 VGPRs/lane ----
    v8f acc[8];
#pragma unroll
    for (int t = 0; t < 8; ++t)
#pragma unroll
        for (int i = 0; i < 8; ++i) acc[t][i] = 0.0f;

    stage(0, 0);

    for (int kc = 0; kc < DIM / 32; ++kc) {
        __syncthreads();                       // staging of buf kc&1 complete
        const int buf = kc & 1;
        if (kc + 1 < DIM / 32) stage(buf ^ 1, (kc + 1) * 32);

        const int k0 = kc * 32;

        // B fragment: B[k,n] = o_bf16[aBase+n, k0+k]
        // 16-bit B layout: lane = column n (=lane&15); lanes 0-15 hold K 0..7 | 16..23,
        // lanes 16-31 hold K 8..15 | 24..31 (same order as A).
        const unsigned short* bp =
            o_bf16 + (size_t)(aBase + m) * DIM + k0;
        V16 bfrag;
        bfrag.h[0] = *(const v8u*)(bp + hi * 8);
        bfrag.h[1] = *(const v8u*)(bp + 16 + hi * 8);

#pragma unroll
        for (int t = 0; t < 8; ++t) {
            // A fragment: A[m,k] = x_bf16[cBase + t*16 + m, k0 + k]
            // 16-bit A layout: lanes 0-15: halves = K 0..7 | 16..23,
            //                  lanes 16-31: halves = K 8..15 | 24..31.
            const unsigned short* rowp = &ldsA[buf][t * 16 + m][0];
            V16 afrag;
            afrag.h[0] = *(const v8u*)(rowp + hi * 8);        // ds_load_b128
            afrag.h[1] = *(const v8u*)(rowp + 16 + hi * 8);   // ds_load_b128

            acc[t] = __builtin_amdgcn_wmma_f32_16x16x32_bf16(
                         false, afrag.bf, false, bfrag.bf,
                         (short)0, acc[t], false, false);
        }
    }

    // ---- epilogue: trace contribution = sum_c o_f32[a,c] * W[c,a] ----
    // C/D layout: lane holds column a = aBase+m; VGPR v holds row v + 8*hi.
    const int a_col = aBase + m;
    float partial = 0.0f;
#pragma unroll
    for (int t = 0; t < 8; ++t) {
        const float* wp = o_f32 + (size_t)a_col * DIM + cBase + t * 16 + hi * 8;
        v4f w0 = *(const v4f*)(wp);
        v4f w1 = *(const v4f*)(wp + 4);
        partial += w0[0] * acc[t][0] + w0[1] * acc[t][1]
                 + w0[2] * acc[t][2] + w0[3] * acc[t][3]
                 + w1[0] * acc[t][4] + w1[1] * acc[t][5]
                 + w1[2] * acc[t][6] + w1[3] * acc[t][7];
    }
    // fold rows 0-7 with rows 8-15 (lane ^ 16 holds the other half)
    partial += __shfl_xor(partial, 16, 32);

    if (hi == 0) {
        atomicAdd(out + (size_t)d0 * TIMES + a_col, partial);  // 16 adds/elem
    }
}

// ---------------------------------------------------------------------------
// Host launcher
// ---------------------------------------------------------------------------
extern "C" void kernel_launch(void* const* d_in, const int* in_sizes, int n_in,
                              void* d_out, int out_size, void* d_ws, size_t ws_size,
                              hipStream_t stream) {
    const float* x   = (const float*)d_in[0];   // (32, 2048, 2048) f32
    const float* org = (const float*)d_in[1];   // (256, 2048) f32
    float* out = (float*)d_out;                 // (32, 256) f32

    float*          o_f32  = (float*)d_ws;                                   // 2 MB
    unsigned short* o_bf16 = (unsigned short*)((char*)d_ws +
                              (size_t)TIMES * DIM * sizeof(float));          // 1 MB

    zero_out_kernel<<<(BATCH * TIMES + 255) / 256, 256, 0, stream>>>(out);
    normalize_kernel<<<TIMES, 256, 0, stream>>>(org, o_f32, o_bf16);

    dim3 grid(DIM / 128, BATCH);   // (16 c-blocks, 32 batches)
    trace_gemm_kernel<<<grid, 512, 0, stream>>>(x, o_f32, o_bf16, out);
}
